// EdgeConditionedGAT_34059090657440
// MI455X (gfx1250) — matrix operand order repaired and checked by reading, not compiled, not run on hardware
//
#include <hip/hip_runtime.h>
#include <hip/hip_bf16.h>

// ---------------------------------------------------------------------------
// Edge-conditioned GATv2 (2 layers + classifier) for MI455X / gfx1250.
//  - Node transforms + classifier: v_wmma_f32_16x16x32_bf16 (f32 accumulate)
//    with weights pre-packed to bf16 WMMA fragments and staged into LDS via
//    global_load_async_to_lds_b128 (ASYNCcnt / s_wait_asynccnt).
//  - Edge attention stage: f32 VALU, LDS-staged weights, atomic segment ops.
// ---------------------------------------------------------------------------

#define NN   50000
#define NE   800000
#define DINC 128
#define EDIM 16
#define NH   8
#define NC   16
#define HC   128      // NH*NC
#define NOUT 64

typedef __attribute__((ext_vector_type(16))) __bf16 v16bf;
typedef __attribute__((ext_vector_type(8)))  float  v8f;

// ------------------------- order-preserving float keys ---------------------
__device__ __forceinline__ unsigned fkey(float x) {
    unsigned u = __float_as_uint(x);
    return (u >> 31) ? ~u : (u | 0x80000000u);
}
__device__ __forceinline__ float funkey(unsigned k) {
    unsigned u = (k >> 31) ? (k & 0x7fffffffu) : ~k;
    return __uint_as_float(u);
}

// ---------------- CDNA5 async global->LDS copy (ASYNCcnt) ------------------
// the LDS destination address comes from a VGPR plus the offset; for a flat LDS pointer the low 32 bits
// are the LDS offset (ISA 10.2 aperture mapping), so truncation is valid.
__device__ __forceinline__ void async_copy_b128(void* lds_ptr, const void* gptr) {
    unsigned loff = (unsigned)(uintptr_t)lds_ptr;
    asm volatile("global_load_async_to_lds_b128 %0, %1, off"
                 :: "v"(loff), "v"(gptr) : "memory");
}
__device__ __forceinline__ void wait_async0() {
    asm volatile("s_wait_asynccnt 0x0" ::: "memory");
}

// ------------------------------ fills --------------------------------------
__global__ void fill_f32(float* p, float v, int n) {
    int i = blockIdx.x * blockDim.x + threadIdx.x;
    if (i < n) p[i] = v;
}
__global__ void fill_u32(unsigned* p, unsigned v, int n) {
    int i = blockIdx.x * blockDim.x + threadIdx.x;
    if (i < n) p[i] = v;
}

// ---------------------------------------------------------------------------
// Pack W (f32, [K x ncols] row-major, ncols = NT*16, K mult of 32) into bf16
// WMMA-B fragments: PK[((kb*NT + t)*32 + lane)*16 + j]
//   = bf16( W[(kb*32 + (lane>>4)*16 + j)*ncols + t*16 + (lane&15)] )
// so each lane's 16 B-fragment elements are 32 contiguous bytes.
// ---------------------------------------------------------------------------
__global__ void pack_w_bf16(const float* __restrict__ W, __bf16* __restrict__ PK,
                            int NT, int K)
{
    const int total = (K / 32) * NT * 32 * 16;
    const int i = blockIdx.x * blockDim.x + threadIdx.x;
    if (i >= total) return;
    const int j    = i & 15;
    const int lane = (i >> 4) & 31;
    const int t    = (i >> 9) % NT;
    const int kb   = (i >> 9) / NT;
    const int half = lane >> 4;
    const int m    = lane & 15;
    const int ncols = NT * 16;
    PK[i] = (__bf16)W[(size_t)(kb * 32 + half * 16 + j) * ncols + t * 16 + m];
}

// ---------------------------------------------------------------------------
// WMMA GEMM: Y = X[K=128] @ W + bias (optionally a second W1/B1/Y1 sharing
// the A fragment).  One wave per 16-row strip, NT 16-col tiles.
// Packed bf16 weights are async-staged into LDS once per block.
// ---------------------------------------------------------------------------
template <int NT, bool DUAL>
__global__ void __launch_bounds__(128)
wmma_gemm_kernel(const float* __restrict__ X,
                 const __bf16* __restrict__ PK0, const float* __restrict__ B0,
                 float* __restrict__ Y0,
                 const __bf16* __restrict__ PK1, const float* __restrict__ B1,
                 float* __restrict__ Y1,
                 int nstrips)
{
    constexpr int K     = 128;
    constexpr int ncols = NT * 16;
    constexpr int TOT0  = (K / 32) * NT * 512;      // bf16 elems per matrix
    __shared__ __align__(16) __bf16 sPK[TOT0 * (DUAL ? 2 : 1)];

    // --- async-stage packed weights into LDS (all waves participate) -------
    for (int i = threadIdx.x * 8; i < TOT0; i += 128 * 8)
        async_copy_b128(&sPK[i], PK0 + i);
    if (DUAL)
        for (int i = threadIdx.x * 8; i < TOT0; i += 128 * 8)
            async_copy_b128(&sPK[TOT0 + i], PK1 + i);
    wait_async0();
    __syncthreads();

    const int wave  = threadIdx.x >> 5;
    const int strip = blockIdx.x * 4 + wave;
    if (strip >= nstrips) return;               // wave-uniform: EXEC stays all-1

    const int lane = threadIdx.x & 31;
    const int half = lane >> 4;                 // 0: lanes 0-15, 1: lanes 16-31
    const int m    = lane & 15;

    v8f acc0[NT] = {};
    v8f acc1[NT] = {};

    const float* xrow = X + (size_t)(strip * 16 + m) * K;

#pragma unroll
    for (int k0 = 0; k0 < K; k0 += 32) {
        const int kb = k0 >> 5;
        // A fragment: 16x32 bf16, M = m in both lane halves.
        // elements 0..15 -> K = k0 + half*16 + j
        const float4* xp = (const float4*)(xrow + k0);
        const float4 f0 = xp[half * 4];
        const float4 f1 = xp[half * 4 + 1];
        const float4 f2 = xp[half * 4 + 2];
        const float4 f3 = xp[half * 4 + 3];
        v16bf a;
        a[0]  = (__bf16)f0.x; a[1]  = (__bf16)f0.y; a[2]  = (__bf16)f0.z; a[3]  = (__bf16)f0.w;
        a[4]  = (__bf16)f1.x; a[5]  = (__bf16)f1.y; a[6]  = (__bf16)f1.z; a[7]  = (__bf16)f1.w;
        a[8]  = (__bf16)f2.x; a[9]  = (__bf16)f2.y; a[10] = (__bf16)f2.z; a[11] = (__bf16)f2.w;
        a[12] = (__bf16)f3.x; a[13] = (__bf16)f3.y; a[14] = (__bf16)f3.z; a[15] = (__bf16)f3.w;

#pragma unroll
        for (int t = 0; t < NT; ++t) {
            const __bf16* bp = &sPK[((kb * NT + t) * 32 + lane) * 16];
            const v16bf b = *(const v16bf*)bp;
            acc0[t] = __builtin_amdgcn_wmma_f32_16x16x32_bf16(
                false, a, false, b, (short)0, acc0[t], false, false);
            if (DUAL) {
                const v16bf b1 = *(const v16bf*)(bp + TOT0);
                acc1[t] = __builtin_amdgcn_wmma_f32_16x16x32_bf16(
                    false, a, false, b1, (short)0, acc1[t], false, false);
            }
        }
    }

    // C/D layout: VGPR r -> M = r + half*8, N = m
#pragma unroll
    for (int t = 0; t < NT; ++t) {
        const int col = t * 16 + m;
        const float bias0 = B0[col];
#pragma unroll
        for (int r = 0; r < 8; ++r) {
            const int row = strip * 16 + half * 8 + r;
            Y0[(size_t)row * ncols + col] = acc0[t][r] + bias0;
        }
        if (DUAL) {
            const float bias1 = B1[col];
#pragma unroll
            for (int r = 0; r < 8; ++r) {
                const int row = strip * 16 + half * 8 + r;
                Y1[(size_t)row * ncols + col] = acc1[t][r] + bias1;
            }
        }
    }
}

// ---------------------------------------------------------------------------
// Edge stage.  One thread per (edge, head) pair; idx = e*NH + h.
// ---------------------------------------------------------------------------
__global__ void __launch_bounds__(256)
edge_logits_kernel(const int* __restrict__ src, const int* __restrict__ dst,
                   const float* __restrict__ eattr,
                   const float* __restrict__ We, const float* __restrict__ att,
                   const float* __restrict__ xl, const float* __restrict__ xr,
                   float* __restrict__ logits, unsigned* __restrict__ nodemax)
{
    __shared__ float sWe[EDIM * HC];
    __shared__ float sAtt[HC];
    for (int i = threadIdx.x; i < EDIM * HC; i += blockDim.x) sWe[i] = We[i];
    for (int i = threadIdx.x; i < HC; i += blockDim.x) sAtt[i] = att[i];
    __syncthreads();

    const int idx = blockIdx.x * blockDim.x + threadIdx.x;
    if (idx >= NE * NH) return;
    const int e = idx >> 3;
    const int h = idx & 7;
    const int s = src[e];
    const int d = dst[e];

    float ea[EDIM];
#pragma unroll
    for (int q = 0; q < EDIM; ++q) ea[q] = eattr[(size_t)e * EDIM + q];

    const float* xls = xl + (size_t)s * HC + h * NC;
    const float* xrd = xr + (size_t)d * HC + h * NC;

    float lg = 0.f;
#pragma unroll
    for (int c = 0; c < NC; ++c) {
        float ee = 0.f;
#pragma unroll
        for (int q = 0; q < EDIM; ++q) ee += ea[q] * sWe[q * HC + h * NC + c];
        float mv = xls[c] + xrd[c] + ee;
        mv = fmaxf(mv, 0.2f * mv);              // leaky_relu(0.2)
        lg += mv * sAtt[h * NC + c];
    }
    logits[idx] = lg;
    atomicMax(&nodemax[d * NH + h], fkey(lg));  // deterministic segment max
}

__global__ void __launch_bounds__(256)
edge_expsum_kernel(const int* __restrict__ dst,
                   float* __restrict__ logits,          // in: logit, out: exp
                   const unsigned* __restrict__ nodemax,
                   float* __restrict__ denom)
{
    const int idx = blockIdx.x * blockDim.x + threadIdx.x;
    if (idx >= NE * NH) return;
    const int e = idx >> 3;
    const int h = idx & 7;
    const int d = dst[e];
    const float mx = funkey(nodemax[d * NH + h]);
    const float ex = __expf(logits[idx] - mx);
    logits[idx] = ex;
    atomicAdd(&denom[d * NH + h], ex);
}

__global__ void __launch_bounds__(256)
edge_aggregate_kernel(const int* __restrict__ src, const int* __restrict__ dst,
                      const float* __restrict__ ex, const float* __restrict__ denom,
                      const float* __restrict__ xl, float* __restrict__ agg)
{
    const int idx = blockIdx.x * blockDim.x + threadIdx.x;
    if (idx >= NE * NH) return;
    const int e = idx >> 3;
    const int h = idx & 7;
    const int s = src[e];
    const int d = dst[e];
    const float alpha = ex[idx] / (denom[d * NH + h] + 1e-16f);
    const float* xs = xl + (size_t)s * HC + h * NC;
    float* ad = agg + (size_t)d * HC + h * NC;
#pragma unroll
    for (int c = 0; c < NC; ++c) atomicAdd(&ad[c], alpha * xs[c]);
}

__global__ void __launch_bounds__(256)
node_elu_bias_kernel(float* __restrict__ h, const float* __restrict__ bias, int n)
{
    const int i = blockIdx.x * blockDim.x + threadIdx.x;
    if (i >= n) return;
    const float v = h[i] + bias[i & (HC - 1)];
    h[i] = v > 0.f ? v : (__expf(v) - 1.f);     // ELU
}

// ---------------------------------------------------------------------------
extern "C" void kernel_launch(void* const* d_in, const int* in_sizes, int n_in,
                              void* d_out, int out_size, void* d_ws, size_t ws_size,
                              hipStream_t stream)
{
    const float* x     = (const float*)d_in[0];
    const int*   eidx  = (const int*)  d_in[1];
    const float* eattr = (const float*)d_in[2];
    const float* Wl1   = (const float*)d_in[3];
    const float* bl1   = (const float*)d_in[4];
    const float* Wr1   = (const float*)d_in[5];
    const float* br1   = (const float*)d_in[6];
    const float* We1   = (const float*)d_in[7];
    const float* att1  = (const float*)d_in[8];
    const float* b1    = (const float*)d_in[9];
    const float* Wl2   = (const float*)d_in[10];
    const float* bl2   = (const float*)d_in[11];
    const float* Wr2   = (const float*)d_in[12];
    const float* br2   = (const float*)d_in[13];
    const float* We2   = (const float*)d_in[14];
    const float* att2  = (const float*)d_in[15];
    const float* b2    = (const float*)d_in[16];
    const float* Wc    = (const float*)d_in[17];
    const float* bc    = (const float*)d_in[18];

    const int* src = eidx;
    const int* dst = eidx + NE;

    // Workspace layout (floats). ~26.4M floats + 144KB packed weights.
    const size_t NF = (size_t)NN * HC;          // 6.4M
    float*    ws   = (float*)d_ws;
    float*    xl   = ws;                         // [NN, HC]
    float*    xr   = ws + NF;                    // [NN, HC]
    float*    hb   = ws + 2 * NF;                // [NN, HC]  layer out / agg
    float*    lg   = ws + 3 * NF;                // [NE, NH]  logits -> exp
    float*    den  = ws + 3 * NF + (size_t)NE * NH;           // [NN, NH]
    unsigned* nmax = (unsigned*)(den + (size_t)NN * NH);      // [NN, NH]
    __bf16*   pkbase = (__bf16*)(den + 2 * (size_t)NN * NH);  // 16B aligned
    const int PKW = 16384;                       // bf16 elems per 128x128 pack
    __bf16* pWl1 = pkbase;
    __bf16* pWr1 = pkbase + PKW;
    __bf16* pWl2 = pkbase + 2 * PKW;
    __bf16* pWr2 = pkbase + 3 * PKW;
    __bf16* pWc  = pkbase + 4 * PKW;             // 8192 elems (128x64)

    const int nstrips = NN / 16;                 // 3125
    const dim3 gemmGrid((nstrips + 3) / 4);      // 4 waves/block
    const dim3 gemmBlk(128);
    const int ehThreads = NE * NH;               // 6.4M
    const dim3 ehGrid((ehThreads + 255) / 256), ehBlk(256);
    const dim3 nhGrid((NN * NH + 255) / 256);
    const dim3 nfGrid(((int)NF + 255) / 256);

    // ---- pack all weights to bf16 WMMA fragments (one-shot, tiny) ---------
    pack_w_bf16<<<64, 256, 0, stream>>>(Wl1, pWl1, 8, HC);
    pack_w_bf16<<<64, 256, 0, stream>>>(Wr1, pWr1, 8, HC);
    pack_w_bf16<<<64, 256, 0, stream>>>(Wl2, pWl2, 8, HC);
    pack_w_bf16<<<64, 256, 0, stream>>>(Wr2, pWr2, 8, HC);
    pack_w_bf16<<<32, 256, 0, stream>>>(Wc,  pWc,  4, HC);

    auto layer = [&](const float* xin,
                     const __bf16* pWl, const float* bl,
                     const __bf16* pWr, const float* br,
                     const float* We, const float* att, const float* bias,
                     float* hout) {
        // xl = xin@Wl + bl ; xr = xin@Wr + br   (shared A fragment)
        wmma_gemm_kernel<8, true><<<gemmGrid, gemmBlk, 0, stream>>>(
            xin, pWl, bl, xl, pWr, br, xr, nstrips);
        fill_f32<<<nhGrid, 256, 0, stream>>>(den, 0.f, NN * NH);
        fill_u32<<<nhGrid, 256, 0, stream>>>(nmax, 0u, NN * NH);
        fill_f32<<<nfGrid, 256, 0, stream>>>(hout, 0.f, (int)NF);
        edge_logits_kernel<<<ehGrid, ehBlk, 0, stream>>>(
            src, dst, eattr, We, att, xl, xr, lg, nmax);
        edge_expsum_kernel<<<ehGrid, ehBlk, 0, stream>>>(dst, lg, nmax, den);
        edge_aggregate_kernel<<<ehGrid, ehBlk, 0, stream>>>(
            src, dst, lg, den, xl, hout);
        node_elu_bias_kernel<<<nfGrid, 256, 0, stream>>>(hout, bias, (int)NF);
    };

    // Layer 1: x -> hb
    layer(x, pWl1, bl1, pWr1, br1, We1, att1, b1, hb);
    // Layer 2: hb -> hb (GEMM consumes hb before it is recycled as agg buffer)
    layer(hb, pWl2, bl2, pWr2, br2, We2, att2, b2, hb);
    // Classifier: d_out = hb @ Wc + bc   (50000 x 64, f32)
    wmma_gemm_kernel<4, false><<<gemmGrid, gemmBlk, 0, stream>>>(
        hb, pWc, bc, (float*)d_out, nullptr, nullptr, nullptr, nstrips);
}
